// EdgeScorer_2482491097615
// MI455X (gfx1250) — matrix operand order repaired and checked by reading, not compiled, not run on hardware
//
#include <hip/hip_runtime.h>

typedef __attribute__((ext_vector_type(16))) _Float16 v16h;
typedef __attribute__((ext_vector_type(8)))  _Float16 v8h;
typedef __attribute__((ext_vector_type(4)))  _Float16 v4h;
typedef __attribute__((ext_vector_type(8)))  float    v8f;
typedef __attribute__((ext_vector_type(4)))  float    v4f;

#define HDIM   128   // h feature dim
#define DDEG   16    // candidates per source node (one WMMA M-tile)
#define KTOP   4
#define HIDW   64    // MLP hidden width -> 4 N-tiles of 16
#define W1PACK (8*4*32*16)   // [kstep=8][ntile=4][lane=32][j=16] f16 = 32KB

static __device__ inline v8f splat8(float x) {
    v8f v;
#pragma unroll
    for (int i = 0; i < 8; ++i) v[i] = x;
    return v;
}

// issue the 16 A-operand gathers for one group (src row + per-row dst rows)
static __device__ inline void load_a(const _Float16* __restrict__ h16,
                                     int g, int dnode, int lhi,
                                     v8h (&lo)[8], v8h (&hi)[8]) {
    const _Float16* srow = h16 + (size_t)g     * HDIM + lhi * 8;
    const _Float16* drow = h16 + (size_t)dnode * HDIM + lhi * 8;
#pragma unroll
    for (int t = 0; t < 4; ++t) {
        lo[t]     = *(const v8h*)(srow + t * 32);        // K = base+0..7
        hi[t]     = *(const v8h*)(srow + t * 32 + 16);   // K = base+16..23
        lo[4 + t] = *(const v8h*)(drow + t * 32);
        hi[4 + t] = *(const v8h*)(drow + t * 32 + 16);
    }
}

// opaque use of every A tile: forces all 16 gathers issued before one wait
static __device__ inline void pin_a(v8h (&lo)[8], v8h (&hi)[8]) {
    asm volatile(""
        : "+v"(lo[0]), "+v"(lo[1]), "+v"(lo[2]), "+v"(lo[3]),
          "+v"(lo[4]), "+v"(lo[5]), "+v"(lo[6]), "+v"(lo[7]),
          "+v"(hi[0]), "+v"(hi[1]), "+v"(hi[2]), "+v"(hi[3]),
          "+v"(hi[4]), "+v"(hi[5]), "+v"(hi[6]), "+v"(hi[7]));
}

// ---- f32 -> f16 conversion of node features ------------------------------
__global__ void cvt_h_kernel(const float* __restrict__ h,
                             _Float16* __restrict__ h16, int n) {
    int i = (blockIdx.x * blockDim.x + threadIdx.x) * 4;
    if (i < n) {
        v4f v = *(const v4f*)(h + i);
        v4h o;
#pragma unroll
        for (int j = 0; j < 4; ++j) o[j] = (_Float16)v[j];
        *(v4h*)(h16 + i) = o;
    }
}

// ---- pre-swizzle W1 (256x64 row-major) into WMMA B-operand lane layout ----
// packed[t], t = ((kstep*4 + ntile)*32 + lane)*16 + j
//   K = kstep*32 + (j>=8 ? 16 : 0) + (lane>=16 ? 8 : 0) + (j&7) ,  N = ntile*16 + (lane&15)
__global__ void pack_w1_kernel(const float* __restrict__ W1,
                               _Float16* __restrict__ w1p) {
    int t = blockIdx.x * blockDim.x + threadIdx.x;
    if (t >= W1PACK) return;
    int j     =  t        & 15;
    int lane  = (t >> 4)  & 31;
    int ntile = (t >> 9)  & 3;
    int kstep =  t >> 11;
    int K = kstep * 32 + ((j >> 3) * 16) + ((lane >> 4) * 8) + (j & 7);
    int n = ntile * 16 + (lane & 15);
    w1p[t] = (_Float16)W1[K * HIDW + n];
}

// ---- fused edge MLP + per-node top-4 -------------------------------------
__global__ void __launch_bounds__(256)
edge_mlp_topk_kernel(const _Float16* __restrict__ h16,
                     const _Float16* __restrict__ w1p,
                     const float* __restrict__ b1,
                     const float* __restrict__ W2,
                     const float* __restrict__ b2,
                     const int*   __restrict__ dst,
                     float* __restrict__ outF,
                     int nGroups /* = N nodes */) {
    __shared__ __align__(32) _Float16 w1s[W1PACK];
    // stage swizzled W1 into LDS once per block; B tiles stream from LDS in
    // the steady state (deep ds_load pipeline ahead of the WMMAs).
    {
        const v8h* s = (const v8h*)w1p;
        v8h*       d = (v8h*)w1s;
        for (int i = threadIdx.x; i < W1PACK / 8; i += blockDim.x) d[i] = s[i];
    }
    __syncthreads();

    const int lane = threadIdx.x & 31;
    const int wid  = threadIdx.x >> 5;
    const int lrow = lane & 15;   // M-row (edge-in-group) / N-col within tile
    const int lhi  = lane >> 4;   // half-wave select

    // per-lane bias / W2 values: n = ntile*16 + lrow
    const float b1v0 = b1[lrow],       b1v1 = b1[16 + lrow];
    const float b1v2 = b1[32 + lrow],  b1v3 = b1[48 + lrow];
    const float w2v0 = W2[lrow],       w2v1 = W2[16 + lrow];
    const float w2v2 = W2[32 + lrow],  w2v3 = W2[48 + lrow];
    const float b2v  = b2[0];
    const int   NK   = nGroups * KTOP;

    const int stride = gridDim.x * (blockDim.x >> 5);
    int g = blockIdx.x * (blockDim.x >> 5) + wid;

    // ---- prologue: three-stage dnode pipeline + first A tiles ------------
    // clamped, unconditional index loads (OOB -> row of group 0, value unused)
    int dn_g  = dst[((g          < nGroups) ? g          : 0) * DDEG + lrow];
    int dn_g1 = dst[((g + stride < nGroups) ? g + stride : 0) * DDEG + lrow];
    v8h alo[8], ahi[8];
    load_a(h16, (g < nGroups ? g : 0), dn_g, lhi, alo, ahi);

    for (; g < nGroups; g += stride) {
        const int g1 = g + stride;
        const int g2 = g1 + stride;
        // prefetch dnode TWO groups ahead: no consumer this iteration, so the
        // pin below only needs loadcnt<=1, not a fresh L2 round-trip.
        const int dn_g2 = dst[((g2 < nGroups) ? g2 : 0) * DDEG + lrow];

        // current group's tiles were issued last iteration -> short wait here
        pin_a(alo, ahi);

        v8f acc0 = splat8(b1v0), acc1 = splat8(b1v1);
        v8f acc2 = splat8(b1v2), acc3 = splat8(b1v3);

#pragma unroll
        for (int ks = 0; ks < 8; ++ks) {
            v16h a = __builtin_shufflevector(alo[ks], ahi[ks],
                     0,1,2,3,4,5,6,7,8,9,10,11,12,13,14,15);
            const _Float16* bp = &w1s[(ks * 4) * 512 + lane * 16];
            acc0 = __builtin_amdgcn_wmma_f32_16x16x32_f16(
                       false, a, false, *(const v16h*)(bp + 0 * 512),
                       (short)0, acc0, false, false);
            acc1 = __builtin_amdgcn_wmma_f32_16x16x32_f16(
                       false, a, false, *(const v16h*)(bp + 1 * 512),
                       (short)0, acc1, false, false);
            acc2 = __builtin_amdgcn_wmma_f32_16x16x32_f16(
                       false, a, false, *(const v16h*)(bp + 2 * 512),
                       (short)0, acc2, false, false);
            acc3 = __builtin_amdgcn_wmma_f32_16x16x32_f16(
                       false, a, false, *(const v16h*)(bp + 3 * 512),
                       (short)0, acc3, false, false);
        }

        // ---- double-buffer: issue NEXT group's 16 gathers with the
        // already-register-resident dn_g1 (loaded one iteration ago).
        v8h nlo[8], nhi[8];
        load_a(h16, (g1 < nGroups ? g1 : g), dn_g1, lhi, nlo, nhi);

        // layer 2: relu + dot with W2 (cross-lane over the 16 N-columns)
        float part[8];
#pragma unroll
        for (int r = 0; r < 8; ++r) {
            float s = fmaxf(acc0[r], 0.f) * w2v0 + fmaxf(acc1[r], 0.f) * w2v1
                    + fmaxf(acc2[r], 0.f) * w2v2 + fmaxf(acc3[r], 0.f) * w2v3;
            s += __shfl_xor(s, 1, 16);
            s += __shfl_xor(s, 2, 16);
            s += __shfl_xor(s, 4, 16);
            s += __shfl_xor(s, 8, 16);
            part[r] = s + b2v;   // logit for edge (r + lhi*8), replicated in half-wave
        }
        // exchange halves so every lane holds all 16 logits
        float oth[8];
#pragma unroll
        for (int r = 0; r < 8; ++r) oth[r] = __shfl_xor(part[r], 16, 32);

        float sc[16];
#pragma unroll
        for (int i = 0; i < 16; ++i)
            sc[i] = ((i >> 3) == lhi) ? part[i & 7] : oth[i & 7];

        // rank-based parallel top-4: lane (lhi==0, lrow=e) owns edge e.
        // Strict total order (value desc, index asc on ties) == jax top_k.
        const float mys = sc[lrow];
        int rank = 0;
#pragma unroll
        for (int j = 0; j < 16; ++j)
            rank += (sc[j] > mys) || ((sc[j] == mys) && (j < lrow));

        if (lhi == 0 && rank < KTOP) {
            const int base = g * KTOP + rank;
            const float w  = 1.0f / (1.0f + __expf(-mys));  // sigmoid (monotone)
            outF[base]          = (float)g;       // edge_index row 0 (src)
            outF[NK + base]     = (float)dn_g;    // edge_index row 1 (dst)
            outF[2 * NK + base] = w;              // edge_w
        }

        // rotate pipelines (phi-coalesced: no real moves expected)
#pragma unroll
        for (int t = 0; t < 8; ++t) { alo[t] = nlo[t]; ahi[t] = nhi[t]; }
        dn_g  = dn_g1;
        dn_g1 = dn_g2;
    }
}

extern "C" void kernel_launch(void* const* d_in, const int* in_sizes, int n_in,
                              void* d_out, int out_size, void* d_ws, size_t ws_size,
                              hipStream_t stream) {
    const float* h  = (const float*)d_in[0];
    const float* W1 = (const float*)d_in[1];
    const float* b1 = (const float*)d_in[2];
    const float* W2 = (const float*)d_in[3];
    const float* b2 = (const float*)d_in[4];
    // d_in[5] = src (implied by uniform degree; unused)
    const int*   dst = (const int*)d_in[6];

    const int n_h     = in_sizes[0];          // N * H
    const int nGroups = n_h / HDIM;           // N source nodes (E/D groups)

    _Float16* h16 = (_Float16*)d_ws;
    _Float16* w1p = (_Float16*)((char*)d_ws + (size_t)n_h * sizeof(_Float16));

    cvt_h_kernel<<<(n_h / 4 + 255) / 256, 256, 0, stream>>>(h, h16, n_h);
    pack_w1_kernel<<<(W1PACK + 255) / 256, 256, 0, stream>>>(W1, w1p);
    edge_mlp_topk_kernel<<<1024, 256, 0, stream>>>(
        h16, w1p, b1, W2, b2, dst, (float*)d_out, nGroups);
}
